// TimeConv_48284022342091
// MI455X (gfx1250) — compile-verified, matrix operand order checked by reading, not run on hardware
//
#include <hip/hip_runtime.h>
#include <hip/hip_bf16.h>

// ---------------------------------------------------------------------------
// TimeConv GNN for MI455X (gfx1250, wave32, WMMA).
// All MLP GEMMs use v_wmma_f32_16x16x32_f16 (f16 in, f32 accumulate).
// Gathers / softmax / attention stay in f32 VALU.
//
// Input-order assumption: setup_inputs() dict insertion order, params nested
// dicts also insertion order (pi, neigh_gate, neigh_mod, type, pos, glob,
// out, attn_m; each MLP as W1,b1,W2,b2).
// ---------------------------------------------------------------------------

constexpr int kL = 8;
constexpr int kP = 16384;
constexpr int kN = kL * kP;
constexpr int kD = 16;
constexpr int kH = 128;
constexpr int kF = 32;
constexpr int kG = kP / 2;

typedef __attribute__((ext_vector_type(16))) _Float16 v16h;
typedef __attribute__((ext_vector_type(8)))  _Float16 v8h;
typedef __attribute__((ext_vector_type(8)))  float    v8f;

__device__ __forceinline__ float lrelu(float x) { return x > 0.f ? x : 0.1f * x; }

// A-fragment (16x32 f16, rows = M) from an LDS tile stored row-major with
// leading dimension ld (halfs, ld % 8 == 0 so chunks stay 16B aligned).
// Layout per CDNA5 ISA 7.12.2: lane<16 -> row=lane, K in {0..7, 16..23};
// lane>=16 -> row=lane-16, K in {8..15, 24..31}.
__device__ __forceinline__ v16h afrag_lds(const _Float16* tile, int ld, int kbase) {
  int lane = threadIdx.x & 31;
  int row  = lane & 15;
  int kh   = (lane >> 4) << 3;
  const _Float16* p = tile + row * ld + kbase + kh;
  v8h lo = *(const v8h*)(p);
  v8h hi = *(const v8h*)(p + 16);
  v16h a;
#pragma unroll
  for (int i = 0; i < 8; ++i) { a[i] = lo[i]; a[8 + i] = hi[i]; }
  return a;
}

// B-fragment (32x16 f16) from a pre-packed global buffer: each lane's 16
// halfs are contiguous (packed by k_pack below).
__device__ __forceinline__ v16h bfrag_g(const _Float16* packed, int tileIdx) {
  int lane = threadIdx.x & 31;
  return *(const v16h*)(packed + ((size_t)tileIdx * 32 + lane) * 16);
}

__device__ __forceinline__ v8f wmma16(v16h a, v16h b, v8f c) {
  return __builtin_amdgcn_wmma_f32_16x16x32_f16(false, a, false, b, (short)0, c,
                                                false, false);
}

// ---------------------------------------------------------------------------
// Pack an f32 weight matrix (Krows x Ncols, row-major) into f16 B-fragment
// tiles of 32x16.  tile = kt*NT + nt; per lane 16 contiguous halfs.
// Zero-pads K beyond Krows (used for mod MLP: 161 -> 192).
// ---------------------------------------------------------------------------
__global__ void k_pack(const float* __restrict__ src, _Float16* __restrict__ dst,
                       int Krows, int Ncols, int NT) {
  int tile = blockIdx.x;
  int nt = tile % NT, kt = tile / NT;
  int lane = threadIdx.x;
  int nl = lane & 15;
  int kh = (lane >> 4) << 3;
  int n = nt * 16 + nl;
  _Float16* d = dst + ((size_t)tile * 32 + lane) * 16;
#pragma unroll
  for (int e = 0; e < 16; ++e) {
    int k = kt * 32 + (e < 8 ? e : e + 8) + kh;
    float v = (k < Krows && n < Ncols) ? src[(size_t)k * Ncols + n] : 0.f;
    d[e] = (_Float16)v;
  }
}

// ---------------------------------------------------------------------------
// h[0:P] = mlp_pi(delay) : 1 -> 64 -> 128 (tiny, VALU)
// ---------------------------------------------------------------------------
__global__ void k_pi(const float* __restrict__ delay,
                     const float* __restrict__ W1, const float* __restrict__ b1,
                     const float* __restrict__ W2, const float* __restrict__ b2,
                     float* __restrict__ h) {
  int p = blockIdx.x * blockDim.x + threadIdx.x;
  if (p >= kP) return;
  float dv = delay[p];
  float hid[64];
#pragma unroll
  for (int j = 0; j < 64; ++j) hid[j] = lrelu(dv * W1[j] + b1[j]);
  float* hr = h + (size_t)p * kH;
  for (int o = 0; o < kH; ++o) {
    float z = b2[o];
#pragma unroll
    for (int j = 0; j < 64; ++j) z += hid[j] * W2[(size_t)j * kH + o];
    hr[o] = z;
  }
}

// ---------------------------------------------------------------------------
// Gate half of one level: gather + per-channel softmax over D, then
// 160 -> 64 -> 128 MLP via WMMA.  Block = 16 nodes, 128 threads (4 waves).
// ---------------------------------------------------------------------------
__global__ void k_gate(int level, float* __restrict__ h,
                       const float* __restrict__ feat,
                       const int* __restrict__ src_gate,
                       const int* __restrict__ is_po,
                       const _Float16* __restrict__ W1p, const float* __restrict__ b1,
                       const _Float16* __restrict__ W2p, const float* __restrict__ b2) {
  constexpr int LDX = 168;  // 160 cols padded, %8==0
  constexpr int LDH = 72;   // 64 cols padded
  __shared__ __align__(16) _Float16 xt[16 * LDX];
  __shared__ __align__(16) _Float16 hid[16 * LDH];

  const int t = threadIdx.x;
  const int gb = blockIdx.x * 16;
  const int prev = (level - 1) * kP;
  const int off = level * kP;
  const float* hp = h + (size_t)prev * kH;

  // Phase A: per node r, channel t: softmax over D=16 messages
  for (int r = 0; r < 16; ++r) {
    const int* sg = src_gate + ((size_t)(level - 1) * kG + gb + r) * kD;
    {
      int c = t;  // 128 channels == 128 threads
      float m[kD];
      float mx = -1e30f;
#pragma unroll
      for (int d = 0; d < kD; ++d) {
        m[d] = hp[(size_t)sg[d] * kH + c];
        mx = fmaxf(mx, m[d]);
      }
      float se = 0.f, sw = 0.f;
#pragma unroll
      for (int d = 0; d < kD; ++d) {
        float e = __expf(m[d] - mx);
        se += e;
        sw += e * m[d];
      }
      xt[r * LDX + c] = (_Float16)(sw / se);
    }
    if (t < kF)
      xt[r * LDX + kH + t] = (_Float16)feat[(size_t)(off + gb + r) * kF + t];
  }
  __syncthreads();

  const int wave = t >> 5;
  const int lane = t & 31;
  const int nl = lane & 15;
  const int moff = (lane >> 4) << 3;

  // GEMM1: (16x160) x (160x64), wave handles one 16-wide N tile
  {
    int nt = wave;
    v8f acc = {};
#pragma unroll
    for (int kt = 0; kt < 5; ++kt)
      acc = wmma16(afrag_lds(xt, LDX, kt * 32), bfrag_g(W1p, kt * 4 + nt), acc);
    int col = nt * 16 + nl;
    float bv = b1[col];
#pragma unroll
    for (int v = 0; v < 8; ++v)
      hid[(v + moff) * LDH + col] = (_Float16)lrelu(acc[v] + bv);
  }
  __syncthreads();

  // GEMM2: (16x64) x (64x128), wave handles two 16-wide N tiles
#pragma unroll
  for (int half = 0; half < 2; ++half) {
    int nt = wave * 2 + half;
    v8f acc = {};
#pragma unroll
    for (int kt = 0; kt < 2; ++kt)
      acc = wmma16(afrag_lds(hid, LDH, kt * 32), bfrag_g(W2p, kt * 8 + nt), acc);
    int col = nt * 16 + nl;
    float bv = b2[col];
#pragma unroll
    for (int v = 0; v < 8; ++v) {
      int node = off + gb + v + moff;
      float x = acc[v] + bv;
      if (is_po[node] != 1) x = fmaxf(x, 0.f);
      h[(size_t)node * kH + col] = x;
    }
  }
}

// ---------------------------------------------------------------------------
// Mod half of one level: attention over D (type/pos MLP dots + hs dot),
// softmax, weighted gather, then 161(->192) -> 64 -> 128 MLP via WMMA.
// ---------------------------------------------------------------------------
__global__ void k_mod(int level, float* __restrict__ h,
                      const float* __restrict__ feat,
                      const float* __restrict__ bitpos,
                      const int* __restrict__ src_mod,
                      const int* __restrict__ is_po,
                      const float* __restrict__ attn,
                      const float* __restrict__ pW1, const float* __restrict__ pb1,
                      const float* __restrict__ pW2, const float* __restrict__ pb2,
                      const float* __restrict__ tW1, const float* __restrict__ tb1,
                      const float* __restrict__ tW2, const float* __restrict__ tb2,
                      const _Float16* __restrict__ W1p, const float* __restrict__ b1,
                      const _Float16* __restrict__ W2p, const float* __restrict__ b2) {
  constexpr int LDX = 200;  // 192 cols padded, %8==0
  constexpr int LDH = 72;
  __shared__ __align__(16) _Float16 xt[16 * LDX];
  __shared__ __align__(16) _Float16 hid[16 * LDH];
  __shared__ float eL[16 * 16];
  __shared__ float aL[16 * 16];

  const int t = threadIdx.x;
  const int gb = blockIdx.x * 16;
  const int prev = (level - 1) * kP;
  const int off = level * kP;
  const float* hp = h + (size_t)prev * kH;
  const size_t sbase = ((size_t)(level - 1) * kG + gb) * kD;

  // Stage 1: per (node r, slot d): e = hs.attn[64:192] + zp(bp).attn[32:64]
  for (int p2 = t; p2 < 256; p2 += 128) {
    int r = p2 >> 4, d = p2 & 15;
    int s = src_mod[sbase + r * kD + d];
    const float* row = hp + (size_t)s * kH;
    float ehs = 0.f;
    for (int c = 0; c < kH; ++c) ehs += row[c] * attn[64 + c];
    float bp = bitpos[sbase + r * kD + d];
    float hr[32];
#pragma unroll
    for (int j = 0; j < 32; ++j) hr[j] = lrelu(bp * pW1[j] + pb1[j]);
    float ezp = 0.f;
    for (int o = 0; o < 32; ++o) {
      float z = pb2[o];
#pragma unroll
      for (int j = 0; j < 32; ++j) z += hr[j] * pW2[j * 32 + o];
      ezp += attn[32 + o] * z;
    }
    eL[r * 16 + d] = ehs + ezp;
  }
  __syncthreads();

  // Stage 2: per node: ezt = zt(fm).attn[0:32], then softmax over d
  if (t < 16) {
    int r = t;
    const float* fm = feat + (size_t)(off + kG + gb + r) * kF;
    float fr[32];
#pragma unroll
    for (int k = 0; k < 32; ++k) fr[k] = fm[k];
    float hr[32];
    for (int j = 0; j < 32; ++j) {
      float x = tb1[j];
#pragma unroll
      for (int k = 0; k < 32; ++k) x += fr[k] * tW1[k * 32 + j];
      hr[j] = lrelu(x);
    }
    float ezt = 0.f;
    for (int o = 0; o < 32; ++o) {
      float z = tb2[o];
#pragma unroll
      for (int j = 0; j < 32; ++j) z += hr[j] * tW2[j * 32 + o];
      ezt += attn[o] * z;
    }
    float ev[16];
    float mx = -1e30f;
#pragma unroll
    for (int d = 0; d < 16; ++d) { ev[d] = eL[r * 16 + d] + ezt; mx = fmaxf(mx, ev[d]); }
    float se = 0.f;
#pragma unroll
    for (int d = 0; d < 16; ++d) { ev[d] = __expf(ev[d] - mx); se += ev[d]; }
    float inv = 1.f / se;
#pragma unroll
    for (int d = 0; d < 16; ++d) aL[r * 16 + d] = ev[d] * inv;
  }
  __syncthreads();

  // Stage 3: build x tile = [alpha-weighted hs (128) | alpha.bp (1) | fm (32) | 0-pad]
  {
    int c = t;  // 128 channels
    for (int r = 0; r < 16; ++r) {
      float acc = 0.f;
#pragma unroll
      for (int d = 0; d < kD; ++d) {
        int s = src_mod[sbase + r * kD + d];
        acc += aL[r * 16 + d] * hp[(size_t)s * kH + c];
      }
      xt[r * LDX + c] = (_Float16)acc;
    }
  }
  if (t < 16) {
    int r = t;
    float acc = 0.f;
#pragma unroll
    for (int d = 0; d < kD; ++d) acc += aL[r * 16 + d] * bitpos[sbase + r * kD + d];
    xt[r * LDX + 128] = (_Float16)acc;
  }
  if (t < kF)
    for (int r = 0; r < 16; ++r)
      xt[r * LDX + 129 + t] = (_Float16)feat[(size_t)(off + kG + gb + r) * kF + t];
  if (t < 31)
    for (int r = 0; r < 16; ++r) xt[r * LDX + 161 + t] = (_Float16)0.f;
  __syncthreads();

  const int wave = t >> 5;
  const int lane = t & 31;
  const int nl = lane & 15;
  const int moff = (lane >> 4) << 3;

  // GEMM1: (16x192) x (192x64)
  {
    int nt = wave;
    v8f acc = {};
#pragma unroll
    for (int kt = 0; kt < 6; ++kt)
      acc = wmma16(afrag_lds(xt, LDX, kt * 32), bfrag_g(W1p, kt * 4 + nt), acc);
    int col = nt * 16 + nl;
    float bv = b1[col];
#pragma unroll
    for (int v = 0; v < 8; ++v)
      hid[(v + moff) * LDH + col] = (_Float16)lrelu(acc[v] + bv);
  }
  __syncthreads();

  // GEMM2: (16x64) x (64x128)
#pragma unroll
  for (int half = 0; half < 2; ++half) {
    int nt = wave * 2 + half;
    v8f acc = {};
#pragma unroll
    for (int kt = 0; kt < 2; ++kt)
      acc = wmma16(afrag_lds(hid, LDH, kt * 32), bfrag_g(W2p, kt * 8 + nt), acc);
    int col = nt * 16 + nl;
    float bv = b2[col];
#pragma unroll
    for (int v = 0; v < 8; ++v) {
      int node = off + kG + gb + v + moff;
      float x = acc[v] + bv;
      if (is_po[node] != 1) x = fmaxf(x, 0.f);
      h[(size_t)node * kH + col] = x;
    }
  }
}

// ---------------------------------------------------------------------------
// Output head: h_glob = mlp_glob(PO_feat) (VALU), then
// [h_gnn | h_glob] (256) -> 128 via WMMA, -> 1 via VALU dot.
// ---------------------------------------------------------------------------
__global__ void k_out(const float* __restrict__ h,
                      const float* __restrict__ POfeat,
                      const float* __restrict__ glW1, const float* __restrict__ glb1,
                      const float* __restrict__ glW2, const float* __restrict__ glb2,
                      const _Float16* __restrict__ W1p, const float* __restrict__ b1,
                      const float* __restrict__ oW2, const float* __restrict__ ob2,
                      float* __restrict__ out) {
  constexpr int LDX = 264;   // 256 cols padded, %8==0
  constexpr int LDHF = 132;  // f32 hidden tile
  __shared__ __align__(16) _Float16 xt[16 * LDX];
  __shared__ float hidG[16 * 64];
  __shared__ float hf[16 * LDHF];

  const int t = threadIdx.x;
  const int pb = blockIdx.x * 16;

  // glob MLP hidden layer: 16 nodes x 64
  for (int idx = t; idx < 16 * 64; idx += 128) {
    int r = idx >> 6, j = idx & 63;
    float pf = POfeat[pb + r];
    hidG[idx] = lrelu(pf * glW1[j] + glb1[j]);
  }
  __syncthreads();

  // x tile: [h_gnn | h_glob]
  for (int r = 0; r < 16; ++r) {
    float z = glb2[t];
#pragma unroll
    for (int j = 0; j < 64; ++j) z += hidG[r * 64 + j] * glW2[(size_t)j * kH + t];
    xt[r * LDX + kH + t] = (_Float16)z;
    xt[r * LDX + t] = (_Float16)h[(size_t)((kL - 1) * kP + pb + r) * kH + t];
  }
  __syncthreads();

  const int wave = t >> 5;
  const int lane = t & 31;
  const int nl = lane & 15;
  const int moff = (lane >> 4) << 3;

  // GEMM: (16x256) x (256x128)
#pragma unroll
  for (int half = 0; half < 2; ++half) {
    int nt = wave * 2 + half;
    v8f acc = {};
#pragma unroll
    for (int kt = 0; kt < 8; ++kt)
      acc = wmma16(afrag_lds(xt, LDX, kt * 32), bfrag_g(W1p, kt * 8 + nt), acc);
    int col = nt * 16 + nl;
    float bv = b1[col];
#pragma unroll
    for (int v = 0; v < 8; ++v)
      hf[(v + moff) * LDHF + col] = lrelu(acc[v] + bv);
  }
  __syncthreads();

  if (t < 16) {
    int r = t;
    float acc = ob2[0];
    for (int c = 0; c < kH; ++c) acc += hf[r * LDHF + c] * oW2[c];
    out[pb + r] = acc;
  }
}

// ---------------------------------------------------------------------------
extern "C" void kernel_launch(void* const* d_in, const int* in_sizes, int n_in,
                              void* d_out, int out_size, void* d_ws, size_t ws_size,
                              hipStream_t stream) {
  (void)in_sizes; (void)n_in; (void)out_size; (void)ws_size;

  const float* delay  = (const float*)d_in[0];
  const float* feat   = (const float*)d_in[1];
  const float* bitpos = (const float*)d_in[2];
  const float* POfeat = (const float*)d_in[3];
  // params (insertion order): pi, neigh_gate, neigh_mod, type, pos, glob, out, attn_m
  const float* piW1 = (const float*)d_in[4];
  const float* pib1 = (const float*)d_in[5];
  const float* piW2 = (const float*)d_in[6];
  const float* pib2 = (const float*)d_in[7];
  const float* gW1  = (const float*)d_in[8];
  const float* gb1  = (const float*)d_in[9];
  const float* gW2  = (const float*)d_in[10];
  const float* gb2  = (const float*)d_in[11];
  const float* mW1  = (const float*)d_in[12];
  const float* mb1  = (const float*)d_in[13];
  const float* mW2  = (const float*)d_in[14];
  const float* mb2  = (const float*)d_in[15];
  const float* tW1  = (const float*)d_in[16];
  const float* tb1  = (const float*)d_in[17];
  const float* tW2  = (const float*)d_in[18];
  const float* tb2  = (const float*)d_in[19];
  const float* pW1  = (const float*)d_in[20];
  const float* pb1  = (const float*)d_in[21];
  const float* pW2  = (const float*)d_in[22];
  const float* pb2  = (const float*)d_in[23];
  const float* glW1 = (const float*)d_in[24];
  const float* glb1 = (const float*)d_in[25];
  const float* glW2 = (const float*)d_in[26];
  const float* glb2 = (const float*)d_in[27];
  const float* oW1  = (const float*)d_in[28];
  const float* ob1  = (const float*)d_in[29];
  const float* oW2  = (const float*)d_in[30];
  const float* ob2  = (const float*)d_in[31];
  const float* attn = (const float*)d_in[32];
  const int* src_gate = (const int*)d_in[33];
  const int* src_mod  = (const int*)d_in[34];
  const int* is_po    = (const int*)d_in[35];

  // workspace: h (N x H f32) then packed f16 weights
  float* h = (float*)d_ws;
  _Float16* wp = (_Float16*)((char*)d_ws + (size_t)kN * kH * sizeof(float));
  _Float16* gW1p = wp;                  // 5x4  tiles
  _Float16* gW2p = gW1p + 20 * 512;     // 2x8  tiles
  _Float16* mW1p = gW2p + 16 * 512;     // 6x4  tiles
  _Float16* mW2p = mW1p + 24 * 512;     // 2x8  tiles
  _Float16* oW1p = mW2p + 16 * 512;     // 8x8  tiles

  k_pack<<<20, 32, 0, stream>>>(gW1, gW1p, 160, 64, 4);
  k_pack<<<16, 32, 0, stream>>>(gW2, gW2p, 64, 128, 8);
  k_pack<<<24, 32, 0, stream>>>(mW1, mW1p, 161, 64, 4);
  k_pack<<<16, 32, 0, stream>>>(mW2, mW2p, 64, 128, 8);
  k_pack<<<64, 32, 0, stream>>>(oW1, oW1p, 256, 128, 8);

  k_pi<<<kP / 256, 256, 0, stream>>>(delay, piW1, pib1, piW2, pib2, h);

  for (int i = 1; i < kL; ++i) {
    k_gate<<<kG / 16, 128, 0, stream>>>(i, h, feat, src_gate, is_po,
                                        gW1p, gb1, gW2p, gb2);
    k_mod<<<kG / 16, 128, 0, stream>>>(i, h, feat, bitpos, src_mod, is_po, attn,
                                       pW1, pb1, pW2, pb2, tW1, tb1, tW2, tb2,
                                       mW1p, mb1, mW2p, mb2);
  }

  k_out<<<kP / 16, 128, 0, stream>>>(h, POfeat, glW1, glb1, glW2, glb2,
                                     oW1p, ob1, oW2, ob2, (float*)d_out);
}
